// BayesianLinear_39711267619209
// MI455X (gfx1250) — compile-verified, matrix operand order checked by reading, not run on hardware
//
#include <hip/hip_runtime.h>
#include <hip/hip_bf16.h>

// Problem sizes (fixed by the reference)
#define IN_F  4096   // K
#define OUT_F 16384  // O
#define NTOK  8192   // N (tokens)

// GEMM tiling: block 128x128 (8 waves, 2x4), wave 64x32, K-slab 64 (2 WMMA K-steps)
#define BM 128
#define BN 128
#define BK 64
#define LDA 72   // LDS row stride (elems): 64 + 8 pad, keeps 16B align (144B rows)
#define LDB 72
#define BUFB (BM * LDA * 2)  // 18432 bytes per LDS buffer (same for A and B tiles)

typedef __attribute__((ext_vector_type(16))) __bf16       v16bf;
typedef __attribute__((ext_vector_type(8)))  float        v8f;
typedef __attribute__((ext_vector_type(4)))  unsigned int v4u;

#if defined(__has_builtin)
#if __has_builtin(__builtin_amdgcn_global_load_async_to_lds_b128) && \
    __has_builtin(__builtin_amdgcn_s_wait_asynccnt)
#define USE_ASYNC_LDS 1
#endif
#endif
#ifndef USE_ASYNC_LDS
#define USE_ASYNC_LDS 0
#endif

// Exact pointee type the async-LDS builtin expects (from clang diagnostic):
//   "__attribute__((__vector_size__(4 * sizeof(int)))) int __device__ *"
typedef __attribute__((vector_size(16))) int v4i_vs;
typedef __attribute__((address_space(1))) v4i_vs* g_v4i_p;  // global (AS1)
typedef __attribute__((address_space(3))) v4i_vs* l_v4i_p;  // LDS (AS3)
#define AS_GLOBAL(p) ((g_v4i_p)(unsigned long long)(p))
#define AS_LDS(p)    ((l_v4i_p)(unsigned int)(unsigned long long)(p))

// ---------------- helpers ----------------

__device__ __forceinline__ unsigned short f2bf(float f) {
  unsigned int u = __float_as_uint(f);
  u += 0x7FFFu + ((u >> 16) & 1u);
  return (unsigned short)(u >> 16);
}

__device__ __forceinline__ float softplus(float r) {
  return (r > 15.0f) ? r : log1pf(__expf(r));
}

// ---------------- prep kernels (elementwise, memory-bound) ----------------

__global__ void prep_weight_kernel(const float4* __restrict__ mu,
                                   const float4* __restrict__ rho,
                                   const float4* __restrict__ eps,
                                   ushort4* __restrict__ wbf) {
  size_t i = (size_t)blockIdx.x * blockDim.x + threadIdx.x;
  float4 m = mu[i], r = rho[i], e = eps[i];
  ushort4 o;
  o.x = f2bf(fmaf(softplus(r.x), e.x, m.x));
  o.y = f2bf(fmaf(softplus(r.y), e.y, m.y));
  o.z = f2bf(fmaf(softplus(r.z), e.z, m.z));
  o.w = f2bf(fmaf(softplus(r.w), e.w, m.w));
  wbf[i] = o;
}

__global__ void prep_x_kernel(const float4* __restrict__ x,
                              ushort4* __restrict__ xbf) {
  size_t i = (size_t)blockIdx.x * blockDim.x + threadIdx.x;
  float4 v = x[i];
  ushort4 o;
  o.x = f2bf(v.x); o.y = f2bf(v.y); o.z = f2bf(v.z); o.w = f2bf(v.w);
  xbf[i] = o;
}

__global__ void prep_bias_kernel(const float* __restrict__ mu,
                                 const float* __restrict__ rho,
                                 const float* __restrict__ eps,
                                 float* __restrict__ b) {
  int i = blockIdx.x * blockDim.x + threadIdx.x;
  b[i] = fmaf(softplus(rho[i]), eps[i], mu[i]);
}

// ---------------- bf16 WMMA GEMM: out[N,O] = Xbf[N,K] @ Wbf[O,K]^T + bias ----------------
// 256 threads = 8 waves (wave32). Block tile 128x128; wave tile 64x32 (4x2 WMMA tiles),
// K-slab 64 = 2 WMMA K-steps per barrier (16 wmma / 24 ds_load per sync).
// Double-buffered LDS selected by a byte offset (no K-loop unroll -> acc regs coalesce),
// fed by ASYNC global->LDS DMA (no data VGPRs, tracked by ASYNCcnt).

__global__ void __launch_bounds__(256, 1)
bayes_gemm_bf16_kernel(const unsigned short* __restrict__ xbf,
                       const unsigned short* __restrict__ wbf,
                       const float* __restrict__ biasf,
                       float* __restrict__ out) {
  __shared__ unsigned short sA[2][BM * LDA];  // 2 x 18432 B
  __shared__ unsigned short sB[2][BN * LDB];  // 2 x 18432 B

  const int tid  = threadIdx.x;
  const int lane = tid & 31;
  const int wave = tid >> 5;

  const int nBlocks = OUT_F / BN;                 // 128
  const int bm = blockIdx.x / nBlocks;            // consecutive blocks share the A row-tile
  const int bn = blockIdx.x % nBlocks;
  const size_t mBase = (size_t)bm * BM;
  const size_t nBase = (size_t)bn * BN;

  // wave grid 2x4 -> each wave 64 (m) x 32 (n)
  const int wm = (wave >> 2) * 64;
  const int wn = (wave & 3) * 32;

  const int half = lane >> 4;  // 0: K 0..7 & 16..23 ; 1: K 8..15 & 24..31
  const int lr   = lane & 15;  // row (A: M, B: N) within a 16-tile

  v8f acc[4][2] = {};

  // staging plan: 16B chunks. Tile = 128 rows x 8 chunks = 1024 chunks; 4 per thread
  // (rows tid>>3 + 32q), per-thread issues 4 A + 4 B async DMAs per K-slab.
  const int stRow = tid >> 3;         // 0..31
  const int stCol = (tid & 7) * 8;    // elem offset of 16B chunk (0..56)

  const unsigned short* aG = xbf + mBase * IN_F;
  const unsigned short* bG = wbf + nBase * IN_F;

  // LDS base addresses (buffer 0); buffer select = +bufOff bytes
  const char* aStage0 = (const char*)&sA[0][stRow * LDA + stCol];
  const char* bStage0 = (const char*)&sB[0][stRow * LDB + stCol];
  const char* aFrag0  = (const char*)&sA[0][(wm + lr) * LDA + half * 8];
  const char* bFrag0  = (const char*)&sB[0][(wn + lr) * LDB + half * 8];

#if USE_ASYNC_LDS
  auto issueTile = [&](unsigned off, int k0) {
#pragma unroll
    for (int q = 0; q < 4; ++q) {
      __builtin_amdgcn_global_load_async_to_lds_b128(
          AS_GLOBAL(aG + (size_t)(stRow + 32 * q) * IN_F + k0 + stCol),
          AS_LDS(aStage0 + off + q * (32 * LDA * 2)), 0, 0);
      __builtin_amdgcn_global_load_async_to_lds_b128(
          AS_GLOBAL(bG + (size_t)(stRow + 32 * q) * IN_F + k0 + stCol),
          AS_LDS(bStage0 + off + q * (32 * LDB * 2)), 0, 0);
    }
  };
#else
  v4u ra[4], rb[4];
  auto loadG = [&](int k0) {
#pragma unroll
    for (int q = 0; q < 4; ++q) {
      ra[q] = *(const v4u*)(aG + (size_t)(stRow + 32 * q) * IN_F + k0 + stCol);
      rb[q] = *(const v4u*)(bG + (size_t)(stRow + 32 * q) * IN_F + k0 + stCol);
    }
  };
  auto stage = [&](unsigned off) {
#pragma unroll
    for (int q = 0; q < 4; ++q) {
      *(v4u*)(aStage0 + off + q * (32 * LDA * 2)) = ra[q];
      *(v4u*)(bStage0 + off + q * (32 * LDB * 2)) = rb[q];
    }
  };
#endif

  const int kIters = IN_F / BK;  // 64

#if USE_ASYNC_LDS
  issueTile(0, 0);
#else
  loadG(0);
  stage(0);
#endif

  unsigned bufOff = 0;  // toggles 0 <-> BUFB
#pragma clang loop unroll(disable)
  for (int kt = 0; kt < kIters; ++kt) {
    const bool more = (kt + 1 < kIters);
#if USE_ASYNC_LDS
    __builtin_amdgcn_s_wait_asynccnt(0);  // my DMA into buf `bufOff` landed
    __syncthreads();                      // everyone's DMA landed; prior reads of buf^1 done
    if (more) issueTile(bufOff ^ BUFB, (kt + 1) * BK);
#else
    __syncthreads();
    if (more) loadG((kt + 1) * BK);
#endif

    union Frag { v16bf v; v4u q[2]; };
    const char* pa = aFrag0 + bufOff;
    const char* pb = bFrag0 + bufOff;

#pragma unroll
    for (int kk = 0; kk < 2; ++kk) {     // two 16x16x32 K-steps within the slab
      Frag b[2];
#pragma unroll
      for (int j = 0; j < 2; ++j) {
        b[j].q[0] = *(const v4u*)(pb + j * (16 * LDB * 2) + kk * 64);
        b[j].q[1] = *(const v4u*)(pb + j * (16 * LDB * 2) + kk * 64 + 32);
      }
#pragma unroll
      for (int i = 0; i < 4; ++i) {
        Frag a;
        a.q[0] = *(const v4u*)(pa + i * (16 * LDA * 2) + kk * 64);
        a.q[1] = *(const v4u*)(pa + i * (16 * LDA * 2) + kk * 64 + 32);
#pragma unroll
        for (int j = 0; j < 2; ++j)
          acc[i][j] = __builtin_amdgcn_wmma_f32_16x16x32_bf16(
              /*neg_a=*/false, a.v, /*neg_b=*/false, b[j].v,
              /*c_mod=*/(short)0, acc[i][j], /*reuse_a=*/false, /*reuse_b=*/false);
      }
    }

#if !USE_ASYNC_LDS
    if (more) stage(bufOff ^ BUFB);
#endif
    bufOff ^= BUFB;
  }

  // Epilogue: C/D 16x16 f32 layout: VGPR r, lane l -> M = r + (l>=16)*8, N = l%16.
  // Single per-lane base pointer; all stores use immediate offsets (max ~3.6MB < 2^23).
  const float bz0 = biasf[nBase + wn + lr];
  const float bz1 = biasf[nBase + wn + 16 + lr];
  float* outBase = out + (mBase + wm + half * 8) * (size_t)OUT_F + (nBase + wn + lr);
#pragma unroll
  for (int i = 0; i < 4; ++i)
#pragma unroll
    for (int r = 0; r < 8; ++r) {
      outBase[(size_t)(i * 16 + r) * OUT_F]      = acc[i][0][r] + bz0;
      outBase[(size_t)(i * 16 + r) * OUT_F + 16] = acc[i][1][r] + bz1;
    }
}

// ---------------- launcher ----------------
// d_ws layout: [ W_bf16 : 128MB ][ X_bf16 : 64MB ][ bias_f32 : 64KB ]  (needs ~193MB)

extern "C" void kernel_launch(void* const* d_in, const int* in_sizes, int n_in,
                              void* d_out, int out_size, void* d_ws, size_t ws_size,
                              hipStream_t stream) {
  const float* x    = (const float*)d_in[0];
  const float* wmu  = (const float*)d_in[1];
  const float* wrho = (const float*)d_in[2];
  const float* weps = (const float*)d_in[3];
  const float* bmu  = (const float*)d_in[4];
  const float* brho = (const float*)d_in[5];
  const float* beps = (const float*)d_in[6];
  float* out = (float*)d_out;

  unsigned short* wbf = (unsigned short*)d_ws;
  unsigned short* xbf = (unsigned short*)((char*)d_ws + (size_t)OUT_F * IN_F * 2);
  float* biasf = (float*)((char*)d_ws + (size_t)OUT_F * IN_F * 2 + (size_t)NTOK * IN_F * 2);

  {
    size_t nvec = (size_t)OUT_F * IN_F / 4;  // 16M float4 groups
    prep_weight_kernel<<<dim3((unsigned)(nvec / 256)), dim3(256), 0, stream>>>(
        (const float4*)wmu, (const float4*)wrho, (const float4*)weps, (ushort4*)wbf);
  }
  {
    size_t nvec = (size_t)NTOK * IN_F / 4;   // 8M float4 groups
    prep_x_kernel<<<dim3((unsigned)(nvec / 256)), dim3(256), 0, stream>>>(
        (const float4*)x, (ushort4*)xbf);
  }
  prep_bias_kernel<<<dim3(OUT_F / 256), dim3(256), 0, stream>>>(bmu, brho, beps, biasf);

  dim3 grid((NTOK / BM) * (OUT_F / BN));  // 64 * 128 = 8192 blocks
  bayes_gemm_bf16_kernel<<<grid, dim3(256), 0, stream>>>(xbf, wbf, biasf, out);
}
